// TimeMixing_979252543523
// MI455X (gfx1250) — matrix operand order repaired and checked by reading, not compiled, not run on hardware
//
#include <hip/hip_runtime.h>

// ---------------------------------------------------------------------------
// RWKV-4 time mixing for MI455X (gfx1250): bf16 WMMA GEMMs + f32 WKV scan.
// B=8, T=2048, D=1024 hardcoded (matches reference).
// GEMMs use double-buffered LDS with GLOBAL_LOAD_ASYNC_TO_LDS_B128 when the
// toolchain exposes it (ASYNCcnt-tracked DMA, overlapped with WMMA).
// ---------------------------------------------------------------------------

#define RWKV_B 8
#define RWKV_T 2048
#define RWKV_D 1024
#define RWKV_M (RWKV_B * RWKV_T)   // 16384 rows for all GEMMs

typedef __attribute__((ext_vector_type(16))) __bf16 bf16x16;
typedef __attribute__((ext_vector_type(8)))  __bf16 bf16x8;
typedef __attribute__((ext_vector_type(8)))  float  floatx8;

#if defined(__gfx1250__) && __has_builtin(__builtin_amdgcn_global_load_async_to_lds_b128)
#define HAVE_ASYNC_LDS 1
#else
#define HAVE_ASYNC_LDS 0
#endif

#if HAVE_ASYNC_LDS
// Builtin signature (from clang diagnostic): param0 = v4i in addrspace(1)
// (printed as "int __vector(4) __device__ *"), param1 = LDS side, then
// imm offset + imm cpol.
typedef int v4i_t __attribute__((vector_size(16)));
typedef __attribute__((address_space(1))) v4i_t* as1_v4i;
typedef __attribute__((address_space(3))) v4i_t* as3_v4i;

__device__ __forceinline__ void async_cp_b128(const __bf16* g, __bf16* l) {
    __builtin_amdgcn_global_load_async_to_lds_b128(
        (as1_v4i)(v4i_t*)(void*)const_cast<__bf16*>(g),
        (as3_v4i)(v4i_t*)(void*)l,
        /*offset=*/0, /*cpol=*/0);
}
__device__ __forceinline__ void wait_async_lds() {
#if __has_builtin(__builtin_amdgcn_s_wait_asynccnt)
    __builtin_amdgcn_s_wait_asynccnt(0);
#else
    asm volatile("s_wait_asynccnt 0" ::: "memory");
#endif
}
#endif

// -------------------------------------------------------------------
// Kernel 1: fp32 [1024x1024] weight -> bf16 transposed [N][K]
// -------------------------------------------------------------------
__global__ __launch_bounds__(256)
void rwkv_transpose_w_bf16(const float* __restrict__ W, __bf16* __restrict__ Wt) {
    __shared__ float tile[32][33];
    const int bx = blockIdx.x * 32;   // n block
    const int by = blockIdx.y * 32;   // k block
    const int tx = threadIdx.x;       // 0..31
    const int ty = threadIdx.y;       // 0..7
#pragma unroll
    for (int j = 0; j < 32; j += 8)
        tile[ty + j][tx] = W[(size_t)(by + ty + j) * RWKV_D + (bx + tx)];
    __syncthreads();
#pragma unroll
    for (int j = 0; j < 32; j += 8)
        Wt[(size_t)(bx + ty + j) * RWKV_D + (by + tx)] = (__bf16)tile[tx][ty + j];
}

// -------------------------------------------------------------------
// Kernel 2: token-shift mixing -> bf16 xk/xv/xr; also emit x[:, -1, :]
// -------------------------------------------------------------------
__global__ __launch_bounds__(256)
void rwkv_mix_bf16(const float* __restrict__ x, const float* __restrict__ last_x,
                   const float* __restrict__ tmk, const float* __restrict__ tmv,
                   const float* __restrict__ tmr,
                   __bf16* __restrict__ xk, __bf16* __restrict__ xv,
                   __bf16* __restrict__ xr, float* __restrict__ lastx_out) {
    const size_t i = (size_t)blockIdx.x * blockDim.x + threadIdx.x;  // over B*T*D
    const int d  = (int)(i & (RWKV_D - 1));
    const size_t bt = i >> 10;
    const int t = (int)(bt & (RWKV_T - 1));
    const int b = (int)(bt >> 11);
    const float xc = x[i];
    const float xp = (t == 0) ? last_x[(size_t)b * RWKV_D + d] : x[i - RWKV_D];
    const float mk = tmk[d], mv = tmv[d], mr = tmr[d];
    xk[i] = (__bf16)(xc * mk + xp * (1.0f - mk));
    xv[i] = (__bf16)(xc * mv + xp * (1.0f - mv));
    xr[i] = (__bf16)(xc * mr + xp * (1.0f - mr));
    if (t == RWKV_T - 1) lastx_out[(size_t)b * RWKV_D + d] = xc;
}

// -------------------------------------------------------------------
// Kernel 3: bf16 WMMA GEMM  C[M,N] = A[M,K] * Bt[N,K]^T  (f32 accum)
// Block tile 128x128, 8 waves (wave32), each wave 32(M) x 64(N),
// K-step 32 via v_wmma_f32_16x16x32_bf16, double-buffered LDS.
// act==1 -> sigmoid epilogue (v_rcp_f32 based).
// -------------------------------------------------------------------
#define LDS_STRIDE 40  // 32 + 8 pad; 80B rows keep 16B alignment, spread banks

__global__ __launch_bounds__(256)
void rwkv_gemm_bf16(const __bf16* __restrict__ A, const __bf16* __restrict__ Bt,
                    float* __restrict__ C, const int K, const int N, const int act) {
    __shared__ __bf16 lds_a[2][128 * LDS_STRIDE];
    __shared__ __bf16 lds_b[2][128 * LDS_STRIDE];

    const int tid  = threadIdx.x;
    const int lane = tid & 31;
    const int wave = tid >> 5;   // 0..7
    const int wm   = wave >> 1;  // 0..3 : 32-row slice
    const int wn   = wave & 1;   // 0..1 : 64-col slice
    const int m0   = blockIdx.x * 128;
    const int n0   = blockIdx.y * 128;

    floatx8 acc[2][4];
    const floatx8 zero = {};
#pragma unroll
    for (int i = 0; i < 2; ++i)
#pragma unroll
        for (int j = 0; j < 4; ++j) acc[i][j] = zero;

    // cooperative tile loads: 256 threads, 2 threads per row, 16 elems each
    const int lrow = tid >> 1;
    const int lcol = (tid & 1) << 4;
    const __bf16* gA = A  + (size_t)(m0 + lrow) * K + lcol;
    const __bf16* gB = Bt + (size_t)(n0 + lrow) * K + lcol;
    const int soff = lrow * LDS_STRIDE + lcol;

    const int am = lane & 15;
    const int kh = (lane >> 4) << 3;   // A: K-half 0 or 8
    const int kb = (lane >> 4) << 4;   // unused: K-base 0 or 16 (B loads use kh like A)
    const int cn = lane & 15;
    const int cm = (lane >> 4) << 3;

    bf16x16 afrag[2];
    bf16x16 bfrag[4];
    auto compute = [&](int buf) {
        // A fragments: lane holds row m=(lane&15); elems 0..7 = K kh..kh+7,
        // elems 8..15 = K 16+kh..16+kh+7 (ISA 16-bit A 16x32 layout)
#pragma unroll
        for (int i = 0; i < 2; ++i) {
            const int row = wm * 32 + i * 16 + am;
            union { bf16x16 v; bf16x8 h[2]; } u;
            u.h[0] = *(const bf16x8*)&lds_a[buf][row * LDS_STRIDE + kh];
            u.h[1] = *(const bf16x8*)&lds_a[buf][row * LDS_STRIDE + 16 + kh];
            afrag[i] = u.v;
        }
        // B fragments: lane holds column n=(lane&15); elems 0..7 = K kh..kh+7,
        // elems 8..15 = K 16+kh..16+kh+7 (same K order as A)
#pragma unroll
        for (int j = 0; j < 4; ++j) {
            const int col = wn * 64 + j * 16 + am;
            union { bf16x16 v; bf16x8 h[2]; } u;
            u.h[0] = *(const bf16x8*)&lds_b[buf][col * LDS_STRIDE + kh];
            u.h[1] = *(const bf16x8*)&lds_b[buf][col * LDS_STRIDE + 16 + kh];
            bfrag[j] = u.v;
        }
#pragma unroll
        for (int i = 0; i < 2; ++i)
#pragma unroll
            for (int j = 0; j < 4; ++j)
                acc[i][j] = __builtin_amdgcn_wmma_f32_16x16x32_bf16(
                    false, afrag[i], false, bfrag[j], (short)0, acc[i][j],
                    false, false);
    };

    const int ntiles = K >> 5;

#if HAVE_ASYNC_LDS
    // ---- async DMA double-buffer pipeline ----
    async_cp_b128(gA,     &lds_a[0][soff]);
    async_cp_b128(gA + 8, &lds_a[0][soff + 8]);
    async_cp_b128(gB,     &lds_b[0][soff]);
    async_cp_b128(gB + 8, &lds_b[0][soff + 8]);
    wait_async_lds();
    __syncthreads();
    for (int kt = 0; kt < ntiles; ++kt) {
        const int cur = kt & 1;
        if (kt + 1 < ntiles) {   // DMA next tile while we compute this one
            const int ko = (kt + 1) << 5;
            async_cp_b128(gA + ko,     &lds_a[1 - cur][soff]);
            async_cp_b128(gA + ko + 8, &lds_a[1 - cur][soff + 8]);
            async_cp_b128(gB + ko,     &lds_b[1 - cur][soff]);
            async_cp_b128(gB + ko + 8, &lds_b[1 - cur][soff + 8]);
        }
        compute(cur);
        wait_async_lds();
        __syncthreads();
    }
#else
    // ---- register-staged double-buffer pipeline (fallback) ----
    bf16x8 ra0 = *(const bf16x8*)gA;
    bf16x8 ra1 = *(const bf16x8*)(gA + 8);
    bf16x8 rb0 = *(const bf16x8*)gB;
    bf16x8 rb1 = *(const bf16x8*)(gB + 8);
    *(bf16x8*)&lds_a[0][soff]     = ra0;
    *(bf16x8*)&lds_a[0][soff + 8] = ra1;
    *(bf16x8*)&lds_b[0][soff]     = rb0;
    *(bf16x8*)&lds_b[0][soff + 8] = rb1;
    __syncthreads();
    for (int kt = 0; kt < ntiles; ++kt) {
        const int cur = kt & 1;
        const bool more = (kt + 1 < ntiles);
        if (more) {   // issue global loads early; latency hidden by compute
            const int ko = (kt + 1) << 5;
            ra0 = *(const bf16x8*)(gA + ko);
            ra1 = *(const bf16x8*)(gA + ko + 8);
            rb0 = *(const bf16x8*)(gB + ko);
            rb1 = *(const bf16x8*)(gB + ko + 8);
        }
        compute(cur);
        if (more) {
            *(bf16x8*)&lds_a[1 - cur][soff]     = ra0;
            *(bf16x8*)&lds_a[1 - cur][soff + 8] = ra1;
            *(bf16x8*)&lds_b[1 - cur][soff]     = rb0;
            *(bf16x8*)&lds_b[1 - cur][soff + 8] = rb1;
        }
        __syncthreads();
    }
#endif

    // Epilogue: C layout -> lane(0..15): elem e = row e, col=lane;
    // lane(16..31): elem e = row 8+e, col=lane-16.
#pragma unroll
    for (int i = 0; i < 2; ++i) {
#pragma unroll
        for (int j = 0; j < 4; ++j) {
            const int gm = m0 + wm * 32 + i * 16 + cm;
            const int gn = n0 + wn * 64 + j * 16 + cn;
            float* cp = C + (size_t)gm * N + gn;
#pragma unroll
            for (int e = 0; e < 8; ++e) {
                float v = acc[i][j][e];
                if (act) v = __builtin_amdgcn_rcpf(1.0f + __expf(-v));
                cp[(size_t)e * N] = v;
            }
        }
    }
}

// -------------------------------------------------------------------
// Kernel 4: sequential WKV scan, one thread per (b,d) channel.
// t+1 loads are software-pipelined ahead of the dependent exp chain.
// Emits rwkv = sigmoid(r)*wkv as bf16, final aa/bb/pp carries as f32.
// -------------------------------------------------------------------
__global__ __launch_bounds__(256)
void rwkv_wkv_scan(const float* __restrict__ k, const float* __restrict__ v,
                   const float* __restrict__ r,
                   const float* __restrict__ aa0, const float* __restrict__ bb0,
                   const float* __restrict__ pp0,
                   const float* __restrict__ time_first,
                   const float* __restrict__ time_decay,
                   __bf16* __restrict__ rwkv,
                   float* __restrict__ aa_out, float* __restrict__ bb_out,
                   float* __restrict__ pp_out) {
    const int idx = blockIdx.x * blockDim.x + threadIdx.x;  // 0..8191
    const int b = idx >> 10;
    const int d = idx & (RWKV_D - 1);
    float aa = aa0[idx], bb = bb0[idx], pp = pp0[idx];
    const float tf = time_first[d];
    const float td = time_decay[d];
    const size_t base = (size_t)b * RWKV_T * RWKV_D + d;

    float kt = k[base], vt = v[base], rt = r[base];
    for (int t = 0; t < RWKV_T; ++t) {
        const size_t off = base + (size_t)t * RWKV_D;
        float kn = 0.0f, vn = 0.0f, rn = 0.0f;
        if (t + 1 < RWKV_T) {  // prefetch next step's operands
            kn = k[off + RWKV_D];
            vn = v[off + RWKV_D];
            rn = r[off + RWKV_D];
        }
        const float ww = tf + kt;
        const float qq = fmaxf(pp, ww);
        const float e1 = __expf(pp - qq);
        const float e2 = __expf(ww - qq);
        const float wkv = (e1 * aa + e2 * vt) / (e1 * bb + e2);
        rwkv[off] = (__bf16)(rt * wkv);
        const float ww2 = pp + td;
        const float qq2 = fmaxf(ww2, kt);
        const float e1b = __expf(ww2 - qq2);
        const float e2b = __expf(kt - qq2);
        aa = e1b * aa + e2b * vt;
        bb = e1b * bb + e2b;
        pp = qq2;
        kt = kn; vt = vn; rt = rn;
    }
    aa_out[idx] = aa;
    bb_out[idx] = bb;
    pp_out[idx] = pp;
}

// -------------------------------------------------------------------
// Host-side orchestration
// -------------------------------------------------------------------
extern "C" void kernel_launch(void* const* d_in, const int* in_sizes, int n_in,
                              void* d_out, int out_size, void* d_ws, size_t ws_size,
                              hipStream_t stream) {
    const float* x    = (const float*)d_in[0];
    const float* lx   = (const float*)d_in[1];
    const float* aa0  = (const float*)d_in[2];
    const float* bb0  = (const float*)d_in[3];
    const float* pp0  = (const float*)d_in[4];
    const float* tmk  = (const float*)d_in[5];
    const float* tmv  = (const float*)d_in[6];
    const float* tmr  = (const float*)d_in[7];
    const float* tfir = (const float*)d_in[8];
    const float* tdec = (const float*)d_in[9];
    const float* Wk   = (const float*)d_in[10];
    const float* Wv   = (const float*)d_in[11];
    const float* Wr   = (const float*)d_in[12];
    const float* Wo   = (const float*)d_in[13];

    float* out       = (float*)d_out;                        // [B,T,D]
    float* lastx_out = out + (size_t)RWKV_M * RWKV_D;        // [B,D]
    float* aa_out    = lastx_out + RWKV_B * RWKV_D;
    float* bb_out    = aa_out + RWKV_B * RWKV_D;
    float* pp_out    = bb_out + RWKV_B * RWKV_D;

    // workspace layout
    char* ws = (char*)d_ws;
    const size_t actBytes = (size_t)RWKV_M * RWKV_D * sizeof(__bf16);  // 32 MB
    const size_t wBytes   = (size_t)RWKV_D * RWKV_D * sizeof(__bf16);  //  2 MB
    const size_t f32Bytes = (size_t)RWKV_M * RWKV_D * sizeof(float);   // 64 MB
    __bf16* XK  = (__bf16*)(ws);                 // reused for r*wkv later
    __bf16* XV  = (__bf16*)(ws + actBytes);
    __bf16* XR  = (__bf16*)(ws + 2 * actBytes);
    __bf16* WkT = (__bf16*)(ws + 3 * actBytes);
    __bf16* WvT = (__bf16*)(ws + 3 * actBytes + wBytes);
    __bf16* WrT = (__bf16*)(ws + 3 * actBytes + 2 * wBytes);
    __bf16* WoT = (__bf16*)(ws + 3 * actBytes + 3 * wBytes);
    float*  Kf  = (float*)(ws + 3 * actBytes + 4 * wBytes);
    float*  Vf  = (float*)(ws + 3 * actBytes + 4 * wBytes + f32Bytes);
    float*  Rf  = (float*)(ws + 3 * actBytes + 4 * wBytes + 2 * f32Bytes);

    // 1) weights -> bf16 transposed
    {
        dim3 blk(32, 8), grd(RWKV_D / 32, RWKV_D / 32);
        rwkv_transpose_w_bf16<<<grd, blk, 0, stream>>>(Wk, WkT);
        rwkv_transpose_w_bf16<<<grd, blk, 0, stream>>>(Wv, WvT);
        rwkv_transpose_w_bf16<<<grd, blk, 0, stream>>>(Wr, WrT);
        rwkv_transpose_w_bf16<<<grd, blk, 0, stream>>>(Wo, WoT);
    }

    // 2) token-shift mix -> bf16 activations (+ last_x output)
    {
        const size_t total = (size_t)RWKV_M * RWKV_D;
        rwkv_mix_bf16<<<(unsigned)(total / 256), 256, 0, stream>>>(
            x, lx, tmk, tmv, tmr, XK, XV, XR, lastx_out);
    }

    // 3) k / v / r projections (r with fused sigmoid)
    {
        dim3 blk(256), grd(RWKV_M / 128, RWKV_D / 128);
        rwkv_gemm_bf16<<<grd, blk, 0, stream>>>(XK, WkT, Kf, RWKV_D, RWKV_D, 0);
        rwkv_gemm_bf16<<<grd, blk, 0, stream>>>(XV, WvT, Vf, RWKV_D, RWKV_D, 0);
        rwkv_gemm_bf16<<<grd, blk, 0, stream>>>(XR, WrT, Rf, RWKV_D, RWKV_D, 1);
    }

    // 4) WKV scan (writes r*wkv into XK buffer as bf16)
    {
        rwkv_wkv_scan<<<(RWKV_B * RWKV_D) / 256, 256, 0, stream>>>(
            Kf, Vf, Rf, aa0, bb0, pp0, tfir, tdec, XK, aa_out, bb_out, pp_out);
    }

    // 5) output projection straight into d_out
    {
        dim3 blk(256), grd(RWKV_M / 128, RWKV_D / 128);
        rwkv_gemm_bf16<<<grd, blk, 0, stream>>>(XK, WoT, out, RWKV_D, RWKV_D, 0);
    }

    (void)in_sizes; (void)n_in; (void)out_size; (void)ws_size;
}
